// SingleHeadAttention_34351148433714
// MI455X (gfx1250) — matrix operand order repaired and checked by reading, not compiled, not run on hardware
//
#include <hip/hip_runtime.h>
#include <math.h>

// Shapes from the reference
#define BATCH 8
#define SEQ   4096
#define HID   768
#define DH    64

typedef __bf16 bf16;
typedef __attribute__((ext_vector_type(16))) __bf16 v16bf;
typedef __attribute__((ext_vector_type(8)))  __bf16 v8bf;
typedef __attribute__((ext_vector_type(8)))  float  v8f;

__device__ __forceinline__ unsigned short bfbits(float f) {
  unsigned u = __builtin_bit_cast(unsigned, f);
  u += 0x7FFFu + ((u >> 16) & 1u);            // round-to-nearest-even
  return (unsigned short)(u >> 16);
}
__device__ __forceinline__ bf16 f2bf(float f) {
  unsigned short h = bfbits(f);
  return __builtin_bit_cast(bf16, h);
}
// Pack two f32 -> two bf16 in one dword: 2x add + 1x v_perm_b32.
__device__ __forceinline__ unsigned pack2(float a, float b) {
  unsigned ua = __builtin_bit_cast(unsigned, a) + 0x8000u;  // round-half-up
  unsigned ub = __builtin_bit_cast(unsigned, b) + 0x8000u;
  return __builtin_amdgcn_perm(ub, ua, 0x07060302u);  // {b.hi16, a.hi16}
}

// Assemble a 16-element bf16 fragment from two 8-element (16B) chunks.
__device__ __forceinline__ v16bf frag_from2(const bf16* lo, const bf16* hi) {
  v8bf a = *(const v8bf*)lo;
  v8bf b = *(const v8bf*)hi;
  v16bf r;
#pragma unroll
  for (int i = 0; i < 8; ++i) { r[i] = a[i]; r[i + 8] = b[i]; }
  return r;
}

__device__ __forceinline__ v8f wmma_bf16(v16bf a, v16bf b, v8f c) {
  // D(16x16 f32) = A(16x32 bf16) x B(32x16 bf16) + C
  return __builtin_amdgcn_wmma_f32_16x16x32_bf16(
      false, a, false, b, (short)0, c, false, false);
}

// ---------------------------------------------------------------------------
// Kernel 1: fused QKV projection.  x[B,S,H] fp32 -> Qb (scaled 1/8, bf16),
// Kb (bf16), Vt (bf16, transposed to [B][DH][S])
// ---------------------------------------------------------------------------
__global__ __launch_bounds__(256) void proj_kernel(
    const float* __restrict__ x,
    const float* __restrict__ Wk, const float* __restrict__ bk,
    const float* __restrict__ Wq, const float* __restrict__ bq,
    const float* __restrict__ Wv, const float* __restrict__ bv,
    bf16* __restrict__ Qb, bf16* __restrict__ Kb, bf16* __restrict__ Vt) {
  __shared__ float xs[16][HID];                    // 48 KB, 16 tokens
  const int tid = threadIdx.x;
  const size_t t0 = (size_t)blockIdx.x * 16;

  for (int i = tid; i < 16 * HID; i += 256)
    xs[i / HID][i % HID] = x[t0 * HID + i];
  __syncthreads();

  if (tid < 192) {
    const int col = tid & 63;
    const int mat = tid >> 6;                      // 0:k 1:q 2:v
    const float* W = (mat == 0) ? Wk : (mat == 1) ? Wq : Wv;
    const float bias = (mat == 0) ? bk[col] : (mat == 1) ? bq[col] : bv[col];
    float acc[16];
#pragma unroll
    for (int r = 0; r < 16; ++r) acc[r] = bias;

    for (int k = 0; k < HID; k += 4) {
      const float w0 = W[(k + 0) * DH + col];
      const float w1 = W[(k + 1) * DH + col];
      const float w2 = W[(k + 2) * DH + col];
      const float w3 = W[(k + 3) * DH + col];
#pragma unroll
      for (int r = 0; r < 16; ++r) {
        const float4 xv = *(const float4*)&xs[r][k];
        acc[r] += xv.x * w0 + xv.y * w1 + xv.z * w2 + xv.w * w3;
      }
    }

    if (mat == 2) {                                // V: transposed [b][d][s]
      const int b = (int)(t0 / SEQ);
      const int s = (int)(t0 % SEQ);
#pragma unroll
      for (int r = 0; r < 16; ++r)
        Vt[((size_t)b * DH + col) * SEQ + s + r] = f2bf(acc[r]);
    } else {
      bf16* dst = (mat == 1) ? Qb : Kb;
      const float sc = (mat == 1) ? 0.125f : 1.0f; // fold 1/sqrt(64) into Q
#pragma unroll
      for (int r = 0; r < 16; ++r)
        dst[(t0 + r) * DH + col] = f2bf(acc[r] * sc);
    }
  }
}

// ---------------------------------------------------------------------------
// Kernel 2: flash attention, transposed formulation.
//   S^T = K x Q^T   (A = K tile, B = Q^T)     -> softmax rows live in lanes
//   O^T = V^T x P^T (A = V^T tile, B = P^T)
// One wave = 16 query rows; 64 keys per main iter; tail = [full32] + masked32.
// ---------------------------------------------------------------------------
__global__ __launch_bounds__(256) void attn_kernel(
    const bf16* __restrict__ Qb, const bf16* __restrict__ Kb,
    const bf16* __restrict__ Vt, float* __restrict__ out) {
  const int lane = threadIdx.x & 31;
  const int half = lane >> 4;
  const int ln   = lane & 15;                      // query index within tile
  const int wave = threadIdx.x >> 5;
  const int b    = blockIdx.y;
  const int q0   = blockIdx.x * 128 + wave * 16;

  const bf16* Qrow = Qb + (size_t)b * SEQ * DH;
  const bf16* Krow = Kb + (size_t)b * SEQ * DH;
  const bf16* Vtb  = Vt + (size_t)b * DH * SEQ;

  // Q^T as B-fragments: lane col = query ln; elems[0..15] = d = 32c+16h+0..15
  v16bf qbf[2];
#pragma unroll
  for (int c = 0; c < 2; ++c) {
    const bf16* p = Qrow + (size_t)(q0 + ln) * DH + c * 32 + half * 16;
    qbf[c] = frag_from2(p, p + 8);
  }

  v8f o[4];                                        // O^T: d = 16f + r + 8h, col = ln
#pragma unroll
  for (int f = 0; f < 4; ++f)
#pragma unroll
    for (int i = 0; i < 8; ++i) o[f][i] = 0.0f;
  float m = -__builtin_inff();
  float l = 0.0f;

  // S^T tile for 16 keys starting at kt (rows = keys r+8h, col = query ln)
  auto score_tile = [&](int kt) -> v8f {
    v8f acc;
#pragma unroll
    for (int i = 0; i < 8; ++i) acc[i] = 0.0f;
#pragma unroll
    for (int c = 0; c < 2; ++c) {
      // K as A-frag: lane row = key; elems[0..15] = d 32c+16h+0..15
      const bf16* kp = Krow + (size_t)(kt + ln) * DH + c * 32 + half * 16;
      acc = wmma_bf16(frag_from2(kp, kp + 8), qbf[c], acc);
    }
    return acc;
  };

  // P^T C-frags (two 16-key tiles) -> one 32-key B-frag via lane^16 exchange
  auto build_pb = [&](const float (&plo)[8], const float (&phi)[8]) -> v16bf {
    unsigned dlo[4], dhi[4];
#pragma unroll
    for (int j = 0; j < 4; ++j) {
      dlo[j] = pack2(plo[2 * j], plo[2 * j + 1]);
      dhi[j] = pack2(phi[2 * j], phi[2 * j + 1]);
    }
    union { v16bf v; unsigned d[8]; } pb;
#pragma unroll
    for (int j = 0; j < 4; ++j) {
      const unsigned send = half ? dlo[j] : dhi[j];
      const unsigned keep = half ? dhi[j] : dlo[j];
      const unsigned recv = (unsigned)__shfl_xor((int)send, 16, 32);
      pb.d[j]     = half ? recv : keep;            // keys base+16h    .. +7
      pb.d[j + 4] = half ? keep : recv;            // keys base+16h+8  .. +15
    }
    return pb.v;
  };

  // V^T A-frag: rows = d = 16f+ln; elems = 32 keys starting at kv
  auto v_frag = [&](int f, int kv) -> v16bf {
    const bf16* vp = Vtb + (size_t)(f * 16 + ln) * SEQ + kv + half * 16;
    return frag_from2(vp, vp + 8);
  };

  // ---- main: 64 keys per iteration, no masking ----
  auto step64 = [&](int kb) {
    v8f s0 = score_tile(kb),      s1 = score_tile(kb + 16);
    v8f s2 = score_tile(kb + 32), s3 = score_tile(kb + 48);

    float cand = fmaxf(s0[0], s0[1]);
#pragma unroll
    for (int r = 2; r < 8; ++r) cand = fmaxf(cand, s0[r]);
#pragma unroll
    for (int r = 0; r < 8; ++r) cand = fmaxf(cand, s1[r]);
#pragma unroll
    for (int r = 0; r < 8; ++r) cand = fmaxf(cand, s2[r]);
#pragma unroll
    for (int r = 0; r < 8; ++r) cand = fmaxf(cand, s3[r]);
    cand = fmaxf(cand, __shfl_xor(cand, 16, 32));
    const float mn = fmaxf(m, cand);
    const float sc = __expf(m - mn);
    m = mn;

    float pa[8], pb_[8], pc[8], pd[8];
    float lsum = 0.0f;
#pragma unroll
    for (int r = 0; r < 8; ++r) {
      pa[r] = __expf(s0[r] - mn);  pb_[r] = __expf(s1[r] - mn);
      pc[r] = __expf(s2[r] - mn);  pd[r]  = __expf(s3[r] - mn);
      lsum += (pa[r] + pb_[r]) + (pc[r] + pd[r]);
    }
    l = l * sc + lsum;
#pragma unroll
    for (int f = 0; f < 4; ++f)
#pragma unroll
      for (int i = 0; i < 8; ++i) o[f][i] *= sc;

    v16bf pb0 = build_pb(pa, pb_);                 // keys kb    .. kb+31
    v16bf pb1 = build_pb(pc, pd);                  // keys kb+32 .. kb+63
#pragma unroll
    for (int f = 0; f < 4; ++f) {
      o[f] = wmma_bf16(v_frag(f, kb), pb0, o[f]);
      o[f] = wmma_bf16(v_frag(f, kb + 32), pb1, o[f]);
    }
  };

  // ---- tail: 32 keys, optionally causal-masked ----
  auto step32 = [&](int kb, bool masked) {
    v8f s0 = score_tile(kb), s1 = score_tile(kb + 16);
    if (masked) {
      const int qrow = q0 + ln;
#pragma unroll
      for (int r = 0; r < 8; ++r) {
        if (kb + r + 8 * half > qrow)      s0[r] = -__builtin_inff();
        if (kb + 16 + r + 8 * half > qrow) s1[r] = -__builtin_inff();
      }
    }
    float cand = fmaxf(s0[0], s0[1]);
#pragma unroll
    for (int r = 2; r < 8; ++r) cand = fmaxf(cand, s0[r]);
#pragma unroll
    for (int r = 0; r < 8; ++r) cand = fmaxf(cand, s1[r]);
    cand = fmaxf(cand, __shfl_xor(cand, 16, 32));
    const float mn = fmaxf(m, cand);
    const float sc = __expf(m - mn);
    m = mn;

    float pa[8], pb_[8];
    float lsum = 0.0f;
#pragma unroll
    for (int r = 0; r < 8; ++r) {
      pa[r]  = __expf(s0[r] - mn);
      pb_[r] = __expf(s1[r] - mn);
      lsum += pa[r] + pb_[r];
    }
    l = l * sc + lsum;
#pragma unroll
    for (int f = 0; f < 4; ++f)
#pragma unroll
      for (int i = 0; i < 8; ++i) o[f][i] *= sc;

    v16bf pb0 = build_pb(pa, pb_);
#pragma unroll
    for (int f = 0; f < 4; ++f)
      o[f] = wmma_bf16(v_frag(f, kb), pb0, o[f]);
  };

  const int nf32 = q0 >> 5;                        // full (unmasked) 32-blocks
  int kb = 0;
  for (int it = 0; it < (nf32 >> 1); ++it, kb += 64) step64(kb);
  if (nf32 & 1) { step32(kb, false); kb += 32; }
  step32(kb, true);                                // masked diagonal tile

  // ---- finalize: combine lane-pair partial sums, normalize, store ----
  l += __shfl_xor(l, 16, 32);
  const float linv = 1.0f / l;
  float* ob = out + ((size_t)b * SEQ + q0 + ln) * DH;   // contiguous in d
#pragma unroll
  for (int f = 0; f < 4; ++f) {
    float4 lo, hi;
    lo.x = o[f][0] * linv; lo.y = o[f][1] * linv;
    lo.z = o[f][2] * linv; lo.w = o[f][3] * linv;
    hi.x = o[f][4] * linv; hi.y = o[f][5] * linv;
    hi.z = o[f][6] * linv; hi.w = o[f][7] * linv;
    *(float4*)&ob[f * 16 + half * 8 + 0] = lo;
    *(float4*)&ob[f * 16 + half * 8 + 4] = hi;
  }
}

// ---------------------------------------------------------------------------
extern "C" void kernel_launch(void* const* d_in, const int* in_sizes, int n_in,
                              void* d_out, int out_size, void* d_ws, size_t ws_size,
                              hipStream_t stream) {
  const float* x  = (const float*)d_in[0];
  const float* Wk = (const float*)d_in[1];
  const float* bk = (const float*)d_in[2];
  const float* Wq = (const float*)d_in[3];
  const float* bq = (const float*)d_in[4];
  const float* Wv = (const float*)d_in[5];
  const float* bv = (const float*)d_in[6];
  float* out = (float*)d_out;

  char* ws = (char*)d_ws;
  const size_t one = (size_t)BATCH * SEQ * DH * sizeof(bf16);  // 4 MB each
  bf16* Qb = (bf16*)(ws);
  bf16* Kb = (bf16*)(ws + one);
  bf16* Vt = (bf16*)(ws + 2 * one);

  proj_kernel<<<(BATCH * SEQ) / 16, 256, 0, stream>>>(x, Wk, bk, Wq, bq, Wv, bv,
                                                      Qb, Kb, Vt);
  dim3 grid(SEQ / 128, BATCH);
  attn_kernel<<<grid, 256, 0, stream>>>(Qb, Kb, Vt, out);
}
